// Attention_1434519077262
// MI455X (gfx1250) — matrix operand order repaired and checked by reading, not compiled, not run on hardware
//
#include <hip/hip_runtime.h>

// MI455X (gfx1250) wave32 WMMA implementation of softmax-free multi-head attention.
// Algebraic rewrite: z_b = xs_b @ Rt_b + bias, Rt_b = sum_h Wq_h^T (Wk_h (xs^T xs) Wv_h^T) Wo_h^T
// -> ~13 GFLOP of 256/1024-dim GEMMs instead of ~103 GFLOP + 256MB score tensor.
// GEMMs are NT form; operands staged into LDS with double-buffered
// global_load_async_to_lds_b128 (ASYNCcnt) and consumed via ds_load_b128.

typedef __attribute__((ext_vector_type(16))) _Float16 v16h;
typedef __attribute__((ext_vector_type(8)))  _Float16 h8;
typedef __attribute__((ext_vector_type(4)))  _Float16 hv4;
typedef __attribute__((ext_vector_type(8)))  float    v8f;

// ---------------- f32 -> f16 convert (vectorized) ----------------
__global__ __launch_bounds__(256) void cvt_f32_f16(const float* __restrict__ in,
                                                   _Float16* __restrict__ out, int n4) {
  int i = blockIdx.x * blockDim.x + threadIdx.x;
  if (i < n4) {
    float4 v = ((const float4*)in)[i];
    hv4 o;
    o.x = (_Float16)v.x; o.y = (_Float16)v.y; o.z = (_Float16)v.z; o.w = (_Float16)v.w;
    ((hv4*)out)[i] = o;
  }
}

// ---------------- batched tiled f16 transpose via LDS ----------------
__global__ __launch_bounds__(256) void transpose_f16(const _Float16* __restrict__ src,
                                                     _Float16* __restrict__ dst,
                                                     int R, int Cols, int nh,
                                                     size_t dbs, size_t hoff, int dld) {
  __shared__ _Float16 t[32][33];
  int z = blockIdx.z, b = z / nh, h = z - b * nh;
  const _Float16* s = src + (size_t)z * R * Cols;
  _Float16* d = dst + (size_t)b * dbs + (size_t)h * hoff;
  int tx = threadIdx.x, ty = threadIdx.y;
  int r0 = blockIdx.y * 32, c0 = blockIdx.x * 32;
#pragma unroll
  for (int dy = 0; dy < 4; ++dy)
    t[ty + 8 * dy][tx] = s[(size_t)(r0 + ty + 8 * dy) * Cols + c0 + tx];
  __syncthreads();
#pragma unroll
  for (int dy = 0; dy < 4; ++dy)
    d[(size_t)(c0 + ty + 8 * dy) * dld + r0 + tx] = t[tx][ty + 8 * dy];
}

// ---------------- CDNA5 async global->LDS helpers (inline asm, ASYNCcnt) ----------------
__device__ __forceinline__ void async_b128(unsigned lds_off, const _Float16* g) {
  asm volatile("global_load_async_to_lds_b128 %0, %1, off"
               :: "v"(lds_off), "v"(g) : "memory");
}
__device__ __forceinline__ void wait_async0() {
  asm volatile("s_wait_asynccnt 0x0" ::: "memory");
}

// ---------------- NT GEMM, f16 in, f32 accumulate via WMMA ----------------
// C[z] = A[z] * B[z]^T.  A: [M x K] ld=lda, B: [N x K] ld=ldb (row-major f16).
// Block = 256 threads = 8 waves; block tile 128x128; wave tile 64x32 (4x2 WMMA tiles).
// K-slabs of 32 double-buffered in LDS via async loads; rows padded to 40 halves (80B)
// so 16-lane ds_load_b128 fragment reads land on 16 distinct bank groups.
#define LROW 40
__global__ __launch_bounds__(256) void nt_gemm_wmma(
    const _Float16* __restrict__ A, const _Float16* __restrict__ B,
    void* __restrict__ Cout, const float* __restrict__ bias,
    int M, int N, int K, int lda, int ldb, int ldc,
    int nh, size_t Asb, size_t Ash, size_t Bsb, size_t Bsh, int f32out) {
  __shared__ __align__(16) _Float16 lA[2][128 * LROW];
  __shared__ __align__(16) _Float16 lB[2][128 * LROW];

  int z = blockIdx.z, b = z / nh, h = z - b * nh;
  const _Float16* Ab = A + b * Asb + h * Ash;
  const _Float16* Bb = B + b * Bsb + h * Bsh;

  int tid = threadIdx.x;
  int lane = tid & 31;
  int r = lane & 15;        // row-in-tile (A) / col-in-tile (B,D)
  int g = lane >> 4;        // half-wave group selects K / M sub-ranges per ISA layout
  int w = tid >> 5;
  int mblk = blockIdx.y * 128;
  int nblk = blockIdx.x * 128;
  int mw = (w & 1) * 64;    // wave's M offset inside block tile
  int nw = (w >> 1) * 32;   // wave's N offset inside block tile

  // staging decomposition: each thread moves 4x16B per slab (2 A rows + 2 B rows)
  int srow = tid >> 2;      // 0..63
  int sseg = tid & 3;       // 16B segment within 64B row slab

  auto issue = [&](int kk, int buf) {
    const _Float16* ga0 = Ab + (size_t)(mblk + srow) * lda + kk + sseg * 8;
    const _Float16* ga1 = Ab + (size_t)(mblk + srow + 64) * lda + kk + sseg * 8;
    async_b128((unsigned)(uintptr_t)&lA[buf][srow * LROW + sseg * 8], ga0);
    async_b128((unsigned)(uintptr_t)&lA[buf][(srow + 64) * LROW + sseg * 8], ga1);
    const _Float16* gb0 = Bb + (size_t)(nblk + srow) * ldb + kk + sseg * 8;
    const _Float16* gb1 = Bb + (size_t)(nblk + srow + 64) * ldb + kk + sseg * 8;
    async_b128((unsigned)(uintptr_t)&lB[buf][srow * LROW + sseg * 8], gb0);
    async_b128((unsigned)(uintptr_t)&lB[buf][(srow + 64) * LROW + sseg * 8], gb1);
  };

  v8f acc[4][2] = {};
  int nk = K / 32;
  issue(0, 0);

  for (int t = 0; t < nk; ++t) {
    wait_async0();        // slab t fully in LDS (per-wave ASYNCcnt)
    __syncthreads();      // all waves see slab t; all done reading buf being reused next
    if (t + 1 < nk) issue((t + 1) * 32, (t + 1) & 1);  // overlap with compute(t)

    const _Float16* la = lA[t & 1];
    const _Float16* lb = lB[t & 1];
    v16h af[4], bf[2];
#pragma unroll
    for (int i = 0; i < 4; ++i) {
      // A 16x32 f16 fragment: element e -> K = 8g + e + (e>=8 ? 8 : 0)
      const _Float16* ar = la + (mw + i * 16 + r) * LROW + 8 * g;
      h8 lo = *(const h8*)ar;
      h8 hi = *(const h8*)(ar + 16);
      af[i] = __builtin_shufflevector(lo, hi, 0, 1, 2, 3, 4, 5, 6, 7,
                                      8, 9, 10, 11, 12, 13, 14, 15);
    }
#pragma unroll
    for (int j = 0; j < 2; ++j) {
      // B 32x16 f16 fragment: element e -> K = 8g + e + (e>=8 ? 8 : 0)
      const _Float16* br = lb + (nw + j * 16 + r) * LROW + 8 * g;
      h8 lo = *(const h8*)br;
      h8 hi = *(const h8*)(br + 16);
      bf[j] = __builtin_shufflevector(lo, hi, 0, 1, 2, 3, 4, 5, 6, 7,
                                      8, 9, 10, 11, 12, 13, 14, 15);
    }
#pragma unroll
    for (int i = 0; i < 4; ++i)
#pragma unroll
      for (int j = 0; j < 2; ++j)
        acc[i][j] = __builtin_amdgcn_wmma_f32_16x16x32_f16(
            false, af[i], false, bf[j], (short)0, acc[i][j], false, false);
  }

  size_t cstride = (size_t)M * N;
  if (f32out) {
    float* Cp = (float*)Cout + (size_t)z * cstride;
#pragma unroll
    for (int i = 0; i < 4; ++i)
#pragma unroll
      for (int j = 0; j < 2; ++j)
#pragma unroll
        for (int e = 0; e < 8; ++e) {
          int row = mblk + mw + i * 16 + e + 8 * g;  // D layout: VGPR e -> M = e + 8g
          int col = nblk + nw + j * 16 + r;
          Cp[(size_t)row * ldc + col] = acc[i][j][e] + bias[row];
        }
  } else {
    _Float16* Cp = (_Float16*)Cout + (size_t)z * cstride;
#pragma unroll
    for (int i = 0; i < 4; ++i)
#pragma unroll
      for (int j = 0; j < 2; ++j)
#pragma unroll
        for (int e = 0; e < 8; ++e) {
          int row = mblk + mw + i * 16 + e + 8 * g;
          int col = nblk + nw + j * 16 + r;
          Cp[(size_t)row * ldc + col] = (_Float16)acc[i][j][e];
        }
  }
}

extern "C" void kernel_launch(void* const* d_in, const int* in_sizes, int n_in,
                              void* d_out, int out_size, void* d_ws, size_t ws_size,
                              hipStream_t stream) {
  (void)in_sizes; (void)n_in; (void)out_size; (void)ws_size;  // needs ~37 MB of ws
  constexpr int Bn = 16, Cc = 256, Ln = 1024, NH = 4, DH = 256, HD = NH * DH;

  const float* x   = (const float*)d_in[0];   // [B,C,W,H] == xs^T per batch [C,L]
  const float* Wq  = (const float*)d_in[1];   // [H,D,C]
  const float* Wk  = (const float*)d_in[2];
  const float* Wv  = (const float*)d_in[3];
  const float* Wo  = (const float*)d_in[4];   // [C,H*D]
  const float* Wob = (const float*)d_in[5];   // [C]

  constexpr size_t XWT_E = (size_t)Bn * Cc * Ln;
  constexpr size_t W_E   = (size_t)NH * DH * Cc;
  constexpr size_t G_E   = (size_t)Bn * Cc * Cc;
  constexpr size_t BUF_E = (size_t)Bn * NH * DH * Cc;

  char* ws = (char*)d_ws;
  size_t off = 0;
  auto take = [&](size_t elems) {
    void* p = ws + off;
    off = (off + elems * 2 + 255) & ~(size_t)255;
    return p;
  };
  _Float16* xwT  = (_Float16*)take(XWT_E);  // xs^T  [B,C,L] f16
  _Float16* xw   = (_Float16*)take(XWT_E);  // xs    [B,L,C] f16
  _Float16* wq   = (_Float16*)take(W_E);
  _Float16* wk   = (_Float16*)take(W_E);
  _Float16* wv   = (_Float16*)take(W_E);
  _Float16* wqT  = (_Float16*)take(W_E);    // [H,C,D]
  _Float16* wo   = (_Float16*)take(W_E);    // [C,H*D]
  _Float16* gbuf = (_Float16*)take(G_E);    // G, later Rt
  _Float16* bufA = (_Float16*)take(BUF_E);  // T1 -> MT -> STcat
  _Float16* bufB = (_Float16*)take(BUF_E);  // M  -> S  -> RtT

  dim3 tb(256), tt(32, 8);

  // --- convert inputs to f16 ---
  cvt_f32_f16<<<XWT_E / 4 / 256, tb, 0, stream>>>(x, xwT, (int)(XWT_E / 4));
  cvt_f32_f16<<<W_E / 4 / 256, tb, 0, stream>>>(Wq, wq, (int)(W_E / 4));
  cvt_f32_f16<<<W_E / 4 / 256, tb, 0, stream>>>(Wk, wk, (int)(W_E / 4));
  cvt_f32_f16<<<W_E / 4 / 256, tb, 0, stream>>>(Wv, wv, (int)(W_E / 4));
  cvt_f32_f16<<<W_E / 4 / 256, tb, 0, stream>>>(Wo, wo, (int)(W_E / 4));
  // wq[h] [D,C] -> wqT[h] [C,D]
  transpose_f16<<<dim3(8, 8, NH), tt, 0, stream>>>(wq, wqT, DH, Cc, NH, 0, (size_t)Cc * DH, DH);
  // xs^T [C,L] -> xs [L,C]
  transpose_f16<<<dim3(Ln / 32, Cc / 32, Bn), tt, 0, stream>>>(xwT, xw, Cc, Ln, 1,
                                                               (size_t)Ln * Cc, 0, Cc);

  // --- G_b = xs^T xs = NT(xwT, xwT): [C,C], K=L ---
  nt_gemm_wmma<<<dim3(2, 2, Bn), tb, 0, stream>>>(xwT, xwT, gbuf, nullptr,
      Cc, Cc, Ln, Ln, Ln, Cc, 1, (size_t)Cc * Ln, 0, (size_t)Cc * Ln, 0, 0);

  // --- T1_bh = Wk_h @ G_b (G symmetric) -> bufA ---
  nt_gemm_wmma<<<dim3(2, 2, Bn * NH), tb, 0, stream>>>(wk, gbuf, bufA, nullptr,
      DH, Cc, Cc, Cc, Cc, Cc, NH, 0, (size_t)DH * Cc, (size_t)Cc * Cc, 0, 0);

  // --- M_bh = T1 @ Wv_h^T = K^T V -> bufB ---
  nt_gemm_wmma<<<dim3(2, 2, Bn * NH), tb, 0, stream>>>(bufA, wv, bufB, nullptr,
      DH, DH, Cc, Cc, Cc, DH, NH, (size_t)NH * DH * Cc, (size_t)DH * Cc, 0, (size_t)DH * Cc, 0);

  // --- M^T -> bufA ---
  transpose_f16<<<dim3(8, 8, Bn * NH), tt, 0, stream>>>(bufB, bufA, DH, DH, 1,
                                                        (size_t)DH * DH, 0, DH);

  // --- S_bh = M^T @ Wq_h = NT(MT, wqT) -> bufB  (S^T = Wq^T M) ---
  nt_gemm_wmma<<<dim3(2, 2, Bn * NH), tb, 0, stream>>>(bufA, wqT, bufB, nullptr,
      DH, Cc, DH, DH, DH, Cc, NH, (size_t)NH * DH * DH, (size_t)DH * DH, 0, (size_t)Cc * DH, 0);

  // --- STcat[b][c][h*D+d] = S[bh][d][c] -> bufA ([C, H*D] per batch) ---
  transpose_f16<<<dim3(8, 8, Bn * NH), tt, 0, stream>>>(bufB, bufA, DH, Cc, NH,
                                                        (size_t)Cc * HD, DH, HD);

  // --- Rt_b = STcat @ Wo^T = sum_h Wq^T M Wo_h^T -> gbuf ---
  nt_gemm_wmma<<<dim3(2, 2, Bn), tb, 0, stream>>>(bufA, wo, gbuf, nullptr,
      Cc, Cc, HD, HD, HD, Cc, 1, (size_t)Cc * HD, 0, 0, 0, 0);

  // --- Rt^T -> bufB ---
  transpose_f16<<<dim3(8, 8, Bn), tt, 0, stream>>>(gbuf, bufB, Cc, Cc, 1,
                                                   (size_t)Cc * Cc, 0, Cc);

  // --- out[b][c][l] = (Rt^T @ xs^T)[c][l] + bias[c] = NT(RtT, xw), f32 out ---
  nt_gemm_wmma<<<dim3(Ln / 128, 2, Bn), tb, 0, stream>>>(bufB, xw, d_out, Wob,
      Cc, Ln, Cc, Cc, Cc, Ln, 1, (size_t)Cc * Cc, 0, (size_t)Ln * Cc, 0, 1);
}
